// SwinTransformerBlock_53824530153956
// MI455X (gfx1250) — matrix operand order repaired and checked by reading, not compiled, not run on hardware
//
#include <hip/hip_runtime.h>
#include <hip/hip_bf16.h>
#include <math.h>

typedef __attribute__((ext_vector_type(16))) _Float16 v16h;
typedef __attribute__((ext_vector_type(8)))  _Float16 v8h;
typedef __attribute__((ext_vector_type(8)))  float    v8f;

#define TOKENS   50176   // 16 * 56 * 56  (= 1024 windows * 49)
#define DIM      512
#define NHEADS   16
#define HD       32
#define NWIN     1024    // B * 64
#define NTOK     49
#define MLPH     2048
#define QKSCALE  0.17677669529663687f   // 32^-0.5

// ---------------------------------------------------------------------------
// WMMA fragment helpers (CDNA5 16x16x32 f16 layouts, CDNA5 ISA)
// ---------------------------------------------------------------------------
__device__ inline v16h lds_a_frag(const _Float16* A, int ld, int row0, int k0, int lane) {
    // A-matrix 16x32: lane holds row M=lane%16.
    // halves 0..7  -> K = 16*(lane/16) + 0..7
    // halves 8..15 -> K = 16*(lane/16) + 8 + 0..7
    v16h a;
    int m   = row0 + (lane & 15);
    int kb8 = 16 * (lane >> 4);
    const _Float16* p0 = A + (size_t)m * ld + k0 + kb8;
    const _Float16* p1 = p0 + 8;
#pragma unroll
    for (int i = 0; i < 8; ++i) { a[i] = p0[i]; a[i + 8] = p1[i]; }
    return a;
}

__device__ inline v8f wmma_f16(v16h a, v16h b, v8f c) {
    return __builtin_amdgcn_wmma_f32_16x16x32_f16(false, a, false, b, (short)0, c, false, false);
}

// ---------------------------------------------------------------------------
// f32 -> f16 conversion (weights)
// ---------------------------------------------------------------------------
__global__ void cvt_f16_kernel(const float* __restrict__ in, _Float16* __restrict__ out, int n) {
    int i = blockIdx.x * 256 + threadIdx.x;
    if (i < n) out[i] = (_Float16)in[i];
}

// ---------------------------------------------------------------------------
// LN1 + cyclic shift (-3,-3) + window partition -> xw (f16, window-major rows)
// one wave per destination token
// ---------------------------------------------------------------------------
__global__ __launch_bounds__(256) void ln1_shift_kernel(
        const float* __restrict__ x, const float* __restrict__ g,
        const float* __restrict__ b, _Float16* __restrict__ out) {
    int token = blockIdx.x * 8 + (threadIdx.x >> 5);   // window-major dst token
    int lane  = threadIdx.x & 31;
    int w  = token / NTOK, t = token % NTOK;
    int bb = w >> 6, wi = w & 63;
    int wr = wi >> 3, wc = wi & 7;
    int hs  = wr * 7 + t / 7;          // shifted-image coords
    int wsc = wc * 7 + t % 7;
    int h0 = hs + 3;  if (h0 >= 56) h0 -= 56;   // roll(-3): shifted[i] = orig[(i+3)%56]
    int w0 = wsc + 3; if (w0 >= 56) w0 -= 56;
    const float* row = x + ((size_t)bb * 3136 + h0 * 56 + w0) * DIM;
    float s = 0.f, ss = 0.f;
    for (int c = lane; c < DIM; c += 32) { float v = row[c]; s += v; ss += v * v; }
#pragma unroll
    for (int o = 16; o > 0; o >>= 1) { s += __shfl_xor(s, o, 32); ss += __shfl_xor(ss, o, 32); }
    float mean = s * (1.f / DIM);
    float var  = ss * (1.f / DIM) - mean * mean;
    float inv  = rsqrtf(var + 1e-5f);
    _Float16* orow = out + (size_t)token * DIM;
    for (int c = lane; c < DIM; c += 32)
        orow[c] = (_Float16)((row[c] - mean) * inv * g[c] + b[c]);
}

// LN2: plain token order, x2 (f32) -> y (f16)
__global__ __launch_bounds__(256) void ln2_kernel(
        const float* __restrict__ x2, const float* __restrict__ g,
        const float* __restrict__ b, _Float16* __restrict__ out) {
    int token = blockIdx.x * 8 + (threadIdx.x >> 5);
    int lane  = threadIdx.x & 31;
    const float* row = x2 + (size_t)token * DIM;
    float s = 0.f, ss = 0.f;
    for (int c = lane; c < DIM; c += 32) { float v = row[c]; s += v; ss += v * v; }
#pragma unroll
    for (int o = 16; o > 0; o >>= 1) { s += __shfl_xor(s, o, 32); ss += __shfl_xor(ss, o, 32); }
    float mean = s * (1.f / DIM);
    float var  = ss * (1.f / DIM) - mean * mean;
    float inv  = rsqrtf(var + 1e-5f);
    _Float16* orow = out + (size_t)token * DIM;
    for (int c = lane; c < DIM; c += 32)
        orow[c] = (_Float16)((row[c] - mean) * inv * g[c] + b[c]);
}

// ---------------------------------------------------------------------------
// Generic WMMA GEMM: C(MxN) = A(MxK,f16) * B(KxN,f16), fused epilogues.
// Block tile 128x64x32, 256 threads = 8 waves, each wave 32x32 (2x2 WMMA).
// M,N,K all multiples of tile sizes for every call site.
// EPI: 0=QKV scatter (+scale on q), 1=proj (window-reverse+roll+residual, f32)
//      2=GELU -> f16, 3=bias+residual -> f32 (final output)
// ---------------------------------------------------------------------------
template <int EPI>
__global__ __launch_bounds__(256) void gemm_f16_kernel(
        const _Float16* __restrict__ A, const _Float16* __restrict__ Bm,
        int M, int N, int K, const float* __restrict__ bias,
        _Float16* __restrict__ outq, _Float16* __restrict__ outk,
        _Float16* __restrict__ outv, _Float16* __restrict__ outh,
        float* __restrict__ outf, const float* __restrict__ resid) {
    __shared__ __align__(16) _Float16 As[128][48];   // pad: row stride 96B (16B mult)
    __shared__ __align__(16) _Float16 Bs[32][72];    // pad: row stride 144B (16B mult)
    int tid  = threadIdx.x;
    int lane = tid & 31, waveId = tid >> 5;
    int wM = (waveId & 3) * 32, wN = (waveId >> 2) * 32;
    int bm = blockIdx.y * 128, bn = blockIdx.x * 64;

    v8f acc[2][2] = {};

    for (int k0 = 0; k0 < K; k0 += 32) {
        // cooperative load A tile: 128x32 halves = 512 v8h chunks, 2/thread
#pragma unroll
        for (int c = 0; c < 2; ++c) {
            int chunk = tid + c * 256;
            int r = chunk >> 2, kk = (chunk & 3) * 8;
            *(v8h*)&As[r][kk] = *(const v8h*)&A[(size_t)(bm + r) * K + k0 + kk];
        }
        // B tile: 32x64 halves = 256 v8h chunks, 1/thread
        {
            int r = tid >> 3, nn = (tid & 7) * 8;
            *(v8h*)&Bs[r][nn] = *(const v8h*)&Bm[(size_t)(k0 + r) * N + bn + nn];
        }
        __syncthreads();

        v16h af[2], bf[2];
#pragma unroll
        for (int ti = 0; ti < 2; ++ti)
            af[ti] = lds_a_frag(&As[0][0], 48, wM + ti * 16, 0, lane);
#pragma unroll
        for (int tj = 0; tj < 2; ++tj) {
            int n  = wN + tj * 16 + (lane & 15);
            int kb = 16 * (lane >> 4);                 // B: half i -> K = i + 16*(lane/16)
#pragma unroll
            for (int i = 0; i < 16; ++i) bf[tj][i] = Bs[kb + i][n];
        }
#pragma unroll
        for (int ti = 0; ti < 2; ++ti)
#pragma unroll
            for (int tj = 0; tj < 2; ++tj)
                acc[ti][tj] = wmma_f16(af[ti], bf[tj], acc[ti][tj]);
        __syncthreads();
    }

    // epilogue: lane holds col n=lane%16, rows j+8*(lane/16) per C/D layout
#pragma unroll
    for (int ti = 0; ti < 2; ++ti)
#pragma unroll
        for (int tj = 0; tj < 2; ++tj)
#pragma unroll
            for (int j = 0; j < 8; ++j) {
                int m = bm + wM + ti * 16 + j + 8 * (lane >> 4);
                int c = bn + wN + tj * 16 + (lane & 15);
                float val = acc[ti][tj][j] + bias[c];
                if constexpr (EPI == 0) {           // QKV scatter
                    int which = c >> 9, cc = c & 511;
                    int head = cc >> 5, d = cc & 31;
                    int w = m / NTOK, t = m % NTOK;
                    size_t dst = ((size_t)(w * NHEADS + head) * NTOK + t) * HD + d;
                    _Float16 hv = (_Float16)(which == 0 ? val * QKSCALE : val);
                    _Float16* dp = (which == 0) ? outq : (which == 1) ? outk : outv;
                    dp[dst] = hv;
                } else if constexpr (EPI == 1) {    // proj: reverse + roll(+3) + residual
                    int w = m / NTOK, t = m % NTOK;
                    int b2 = w >> 6, wi = w & 63;
                    int wr = wi >> 3, wc = wi & 7;
                    int hs  = wr * 7 + t / 7;
                    int wsc = wc * 7 + t % 7;
                    int h0 = hs + 3;  if (h0 >= 56) h0 -= 56;
                    int w0 = wsc + 3; if (w0 >= 56) w0 -= 56;
                    size_t g = ((size_t)b2 * 3136 + h0 * 56 + w0) * DIM + c;
                    outf[g] = resid[g] + val;
                } else if constexpr (EPI == 2) {    // exact GELU -> f16
                    float ge = 0.5f * val * (1.f + erff(val * 0.7071067811865476f));
                    outh[(size_t)m * N + c] = (_Float16)ge;
                } else {                            // final: bias + residual -> f32 out
                    size_t g = (size_t)m * DIM + c;
                    outf[g] = val + resid[g];
                }
            }
}

// ---------------------------------------------------------------------------
// Attention: one wave per (window, head). 49x49 scores padded to 64x64.
//  S = Q*K^T (16 WMMA) -> +rel-pos bias +shift mask -> softmax (LDS) ->
//  O = P*V (16 WMMA) -> scatter to (token, head*32+d) f16 rows.
// Block = 64 threads = 2 waves; per-wave LDS 28 KB (P aliases Q+K region).
// ---------------------------------------------------------------------------
__device__ inline int shift_region(int a) { return a < 49 ? 0 : (a < 53 ? 1 : 2); }

__global__ __launch_bounds__(64) void attn_kernel(
        const _Float16* __restrict__ Q, const _Float16* __restrict__ Kf,
        const _Float16* __restrict__ V, const float* __restrict__ rpb,
        _Float16* __restrict__ Out) {
    __shared__ __align__(16) char smem[2 * 28672];
    int waveId = threadIdx.x >> 5, lane = threadIdx.x & 31;
    int pair = blockIdx.x * 2 + waveId;          // 0 .. 16383
    int w = pair >> 4, h = pair & 15;

    char* base = smem + waveId * 28672;
    _Float16* qb = (_Float16*)base;              // 64x32
    _Float16* kb = qb + 64 * 32;                 // 64x32
    _Float16* vb = kb + 64 * 32;                 // 64x32
    float*    Sb = (float*)(vb + 64 * 32);       // 64x64 f32
    _Float16* Pb = qb;                           // 64x64 f16, aliases q+k (4096 halves)

    size_t gbase = (size_t)(w * NHEADS + h) * NTOK * HD;
    const _Float16* qg = Q  + gbase;
    const _Float16* kg = Kf + gbase;
    const _Float16* vg = V  + gbase;

    // stage q/k/v, zero-pad rows 49..63
    for (int r = lane; r < 64; r += 32) {
        bool ok = r < NTOK;
        for (int c = 0; c < HD; ++c) {
            int o = r * HD + c;
            qb[o] = ok ? qg[o] : (_Float16)0.f;
            kb[o] = ok ? kg[o] : (_Float16)0.f;
            vb[o] = ok ? vg[o] : (_Float16)0.f;
        }
    }
    // (single-wave: LDS ops are in-order; compiler inserts s_wait_dscnt for frag reads)

    // S = q * k^T : K-dim = HD = 32 -> single WMMA per 16x16 tile
    for (int mi = 0; mi < 4; ++mi) {
        v16h a = lds_a_frag(qb, HD, mi * 16, 0, lane);
#pragma unroll
        for (int ni = 0; ni < 4; ++ni) {
            v16h bf;
            int n  = ni * 16 + (lane & 15);      // key token (row of k)
            int kb0 = 16 * (lane >> 4);
#pragma unroll
            for (int i = 0; i < 16; ++i) bf[i] = kb[n * HD + kb0 + i];
            v8f acc = {};
            acc = wmma_f16(a, bf, acc);
#pragma unroll
            for (int j = 0; j < 8; ++j)
                Sb[(mi * 16 + j + 8 * (lane >> 4)) * 64 + n] = acc[j];
        }
    }

    // bias + mask + softmax; build P (f16) with zero padding
    int wi = w & 63, wr = wi >> 3, wc = wi & 7;
    for (int t = lane; t < 64; t += 32) {
        if (t >= NTOK) {
            for (int j2 = 0; j2 < 64; ++j2) Pb[t * 64 + j2] = (_Float16)0.f;
            continue;
        }
        int yi = t / 7, xi = t % 7;
        int idt = 3 * shift_region(wr * 7 + yi) + shift_region(wc * 7 + xi);
        float mx = -1e30f;
        for (int j2 = 0; j2 < NTOK; ++j2) {
            int yj = j2 / 7, xj = j2 % 7;
            int ridx = (yi - yj + 6) * 13 + (xi - xj + 6);
            float v = Sb[t * 64 + j2] + rpb[ridx * NHEADS + h];
            int idj = 3 * shift_region(wr * 7 + yj) + shift_region(wc * 7 + xj);
            if (idj != idt) v -= 100.f;
            Sb[t * 64 + j2] = v;
            mx = fmaxf(mx, v);
        }
        float sum = 0.f;
        for (int j2 = 0; j2 < NTOK; ++j2) {
            float e = __expf(Sb[t * 64 + j2] - mx);
            Sb[t * 64 + j2] = e;
            sum += e;
        }
        float inv = 1.f / sum;
        for (int j2 = 0; j2 < NTOK; ++j2) Pb[t * 64 + j2] = (_Float16)(Sb[t * 64 + j2] * inv);
        for (int j2 = NTOK; j2 < 64; ++j2) Pb[t * 64 + j2] = (_Float16)0.f;
    }

    // O = P(64x64) * v(64x32): 4 mi x 2 tj tiles, 2 K-steps of 32
    for (int mi = 0; mi < 4; ++mi) {
#pragma unroll
        for (int tj = 0; tj < 2; ++tj) {
            v8f acc = {};
#pragma unroll
            for (int kk = 0; kk < 2; ++kk) {
                v16h a = lds_a_frag(Pb, 64, mi * 16, kk * 32, lane);
                v16h bf;
                int n = tj * 16 + (lane & 15);
                int kbase = kk * 32 + 16 * (lane >> 4);
#pragma unroll
                for (int i = 0; i < 16; ++i) bf[i] = vb[(kbase + i) * HD + n];
                acc = wmma_f16(a, bf, acc);
            }
#pragma unroll
            for (int j = 0; j < 8; ++j) {
                int m = mi * 16 + j + 8 * (lane >> 4);
                if (m < NTOK)
                    Out[((size_t)(w * NTOK + m)) * DIM + h * HD + tj * 16 + (lane & 15)] =
                        (_Float16)acc[j];
            }
        }
    }
}

// ---------------------------------------------------------------------------
// Launch
// ---------------------------------------------------------------------------
extern "C" void kernel_launch(void* const* d_in, const int* in_sizes, int n_in,
                              void* d_out, int out_size, void* d_ws, size_t ws_size,
                              hipStream_t stream) {
    const float* x      = (const float*)d_in[0];
    const float* w_qkv  = (const float*)d_in[1];
    const float* b_qkv  = (const float*)d_in[2];
    const float* rpb    = (const float*)d_in[3];
    const float* w_proj = (const float*)d_in[4];
    const float* b_proj = (const float*)d_in[5];
    const float* g1     = (const float*)d_in[6];
    const float* b1     = (const float*)d_in[7];
    const float* g2     = (const float*)d_in[8];
    const float* b2     = (const float*)d_in[9];
    const float* w_fc1  = (const float*)d_in[10];
    const float* b_fc1  = (const float*)d_in[11];
    const float* w_fc2  = (const float*)d_in[12];
    const float* b_fc2  = (const float*)d_in[13];
    float* out = (float*)d_out;

    char* ws = (char*)d_ws;
    size_t o = 0;
    auto take = [&](size_t bytes) {
        size_t r = o;
        o = (o + bytes + 255) & ~(size_t)255;
        return r;
    };
    _Float16* W1 = (_Float16*)(ws + take((size_t)512 * 1536 * 2));  // w_qkv f16
    _Float16* W2 = (_Float16*)(ws + take((size_t)512 * 512 * 2));   // w_proj f16
    _Float16* W3 = (_Float16*)(ws + take((size_t)512 * 2048 * 2));  // w_fc1 f16
    _Float16* W4 = (_Float16*)(ws + take((size_t)2048 * 512 * 2));  // w_fc2 f16
    _Float16* XW = (_Float16*)(ws + take((size_t)TOKENS * DIM * 2)); // xw / attn_out / y
    _Float16* Qb = (_Float16*)(ws + take((size_t)TOKENS * DIM * 2));
    _Float16* Kb = (_Float16*)(ws + take((size_t)TOKENS * DIM * 2));
    _Float16* Vb = (_Float16*)(ws + take((size_t)TOKENS * DIM * 2));
    float*    X2 = (float*)   (ws + take((size_t)TOKENS * DIM * 4)); // post-attn residual
    _Float16* H1 = (_Float16*)(ws + take((size_t)TOKENS * MLPH * 2)); // GELU activations
    (void)ws_size; (void)in_sizes; (void)n_in; (void)out_size;

    // 1) weights -> f16
    cvt_f16_kernel<<<(512 * 1536 + 255) / 256, 256, 0, stream>>>(w_qkv, W1, 512 * 1536);
    cvt_f16_kernel<<<(512 * 512  + 255) / 256, 256, 0, stream>>>(w_proj, W2, 512 * 512);
    cvt_f16_kernel<<<(512 * 2048 + 255) / 256, 256, 0, stream>>>(w_fc1, W3, 512 * 2048);
    cvt_f16_kernel<<<(2048 * 512 + 255) / 256, 256, 0, stream>>>(w_fc2, W4, 2048 * 512);

    // 2) LN1 + shift + window partition -> XW (f16)
    ln1_shift_kernel<<<TOKENS / 8, 256, 0, stream>>>(x, g1, b1, XW);

    // 3) QKV GEMM (50176x512 x 512x1536), scatter per (window, head), q*scale
    gemm_f16_kernel<0><<<dim3(1536 / 64, TOKENS / 128), 256, 0, stream>>>(
        XW, W1, TOKENS, 1536, 512, b_qkv, Qb, Kb, Vb, nullptr, nullptr, nullptr);

    // 4) windowed attention -> attn_out (reuses XW)
    attn_kernel<<<(NWIN * NHEADS) / 2, 64, 0, stream>>>(Qb, Kb, Vb, rpb, XW);

    // 5) proj GEMM + window reverse + roll + residual -> X2 (f32)
    gemm_f16_kernel<1><<<dim3(512 / 64, TOKENS / 128), 256, 0, stream>>>(
        XW, W2, TOKENS, 512, 512, b_proj, nullptr, nullptr, nullptr, nullptr, X2, x);

    // 6) LN2 -> y (f16, reuses XW)
    ln2_kernel<<<TOKENS / 8, 256, 0, stream>>>(X2, g2, b2, XW);

    // 7) fc1 GEMM + GELU -> H1 (f16)
    gemm_f16_kernel<2><<<dim3(MLPH / 64, TOKENS / 128), 256, 0, stream>>>(
        XW, W3, TOKENS, MLPH, 512, b_fc1, nullptr, nullptr, nullptr, H1, nullptr, nullptr);

    // 8) fc2 GEMM + bias + residual -> d_out (f32)
    gemm_f16_kernel<3><<<dim3(512 / 64, TOKENS / 128), 256, 0, stream>>>(
        H1, W4, TOKENS, 512, MLPH, b_fc2, nullptr, nullptr, nullptr, nullptr, out, X2);
}
